// WaveBlock_2654289789227
// MI455X (gfx1250) — matrix operand order repaired and checked by reading, not compiled, not run on hardware
//
#include <hip/hip_runtime.h>
#include <hip/hip_bf16.h>
#include <math.h>
#include <stdint.h>

// ---------------- problem constants ----------------
#define D_MODEL 1024
#define N_HEADS 16
#define N_WAVES 16
#define D_HEAD  64            // D_MODEL / N_HEADS
#define D_FF    4096
#define BATCH   2
#define SEQ     2048
#define NTOK    (BATCH * SEQ) // 4096

typedef __attribute__((ext_vector_type(16))) __bf16 bf16x16;
typedef __attribute__((ext_vector_type(8)))  __bf16 bf16x8;
typedef __attribute__((ext_vector_type(8)))  float  f32x8;
typedef __attribute__((ext_vector_type(4)))  unsigned int u32x4;
typedef __attribute__((ext_vector_type(8)))  int i32x8;
typedef __attribute__((ext_vector_type(4)))  int i32x4;

#if __has_include(<hip/amd_detail/amd_gfx1250_TDM.h>)
#define ATHENA_TDM_6ARG 1
#endif

// ---------------- WMMA fragment loaders ----------------
// A matrix 16x32 bf16 (row major, leading dim = lda elements):
//   lanes 0-15 : row = lane,    K = 0..15   (contiguous)
//   lanes 16-31: row = lane-16, K = 16..31  (contiguous)
__device__ __forceinline__ bf16x16 load_a_frag(const __bf16* __restrict__ A,
                                               int lda, int m0, int k0, int lane) {
    const int row  = m0 + (lane & 15);
    const int koff = k0 + ((lane >> 4) << 4);     // +16 for hi lane group
    const __bf16* p = A + (size_t)row * lda + koff;
    return *(const bf16x16*)(p);
}

// B matrix 32x16 bf16, stored TRANSPOSED [N, K] row major (ldb = K elements):
//   lanes 0-15 : col = lane,    K = 0..15   (contiguous)
//   lanes 16-31: col = lane-16, K = 16..31  (contiguous)
__device__ __forceinline__ bf16x16 load_b_frag(const __bf16* __restrict__ Bt,
                                               int ldb, int n0, int k0, int lane) {
    const int col  = n0 + (lane & 15);
    const int koff = k0 + ((lane >> 4) << 4);     // +16 for hi lane group
    return *(const bf16x16*)(Bt + (size_t)col * ldb + koff);
}

__device__ __forceinline__ f32x8 wmma_bf16(bf16x16 a, bf16x16 b, f32x8 c) {
    return __builtin_amdgcn_wmma_f32_16x16x32_bf16(false, a, false, b,
                                                   (short)0, c, false, false);
}

// ---------------- Tensor Data Mover: 32x32-element bf16 tile  global -> LDS -----------
// D# per CDNA5 ISA ch.8: group0 = {count/type/lds/global addr}, group1 = dims/strides.
// 2D tile: tile_dim0 = 32 elems (64B rows), tile_dim1 = 32 rows, row stride = K elems.
__device__ __forceinline__ void tdm_tile_32x32(const __bf16* gsrc, __bf16* lds_dst,
                                               int stride_elems) {
    const uint64_t ga = (uint64_t)(uintptr_t)gsrc;
    u32x4 g0;
    g0[0] = 1u;                                        // count=1, user mode
    g0[1] = (uint32_t)(uintptr_t)lds_dst;              // lds_addr (addr[31:0] = LDS offset)
    g0[2] = (uint32_t)ga;                              // global_addr[31:0]
    g0[3] = (uint32_t)((ga >> 32) & 0x01FFFFFFull) | (2u << 30);  // addr[56:32] | type=2
    i32x8 g1;
    g1[0] = 0x00010000;          // workgroup_mask=0, data_size=1 (2 bytes)
    g1[1] = (int)(32u << 16);    // tensor_dim0[15:0]=32  (bits 79:48)
    g1[2] = (int)(32u << 16);    // tensor_dim1[15:0]=32  (bits 111:80)
    g1[3] = (int)(32u << 16);    // tile_dim0=32          (bits 127:112)
    g1[4] = 32;                  // tile_dim1=32          (bits 143:128)
    g1[5] = stride_elems;        // tensor_dim0_stride[31:0] (bits 207:160)
    g1[6] = 0;                   // stride hi / dim1_stride lo
    g1[7] = 0;
    i32x4 z4 = {0, 0, 0, 0};
#ifdef ATHENA_TDM_6ARG
    i32x8 z8 = {0, 0, 0, 0, 0, 0, 0, 0};
    __builtin_amdgcn_tensor_load_to_lds(g0, g1, z4, z4, z8, 0);
#else
    __builtin_amdgcn_tensor_load_to_lds(g0, g1, z4, z4, 0);
#endif
}

// ---------------- weight transpose-convert: W[K,N] f32 -> Wt[N,K] bf16 ----------------
__global__ __launch_bounds__(256) void k_cvtw(const float* __restrict__ W,
                                              __bf16* __restrict__ Wt, int K, int N) {
    int idx = blockIdx.x * 256 + threadIdx.x;
    if (idx >= K * N) return;
    int k = idx / N, n = idx - k * N;
    Wt[(size_t)n * K + k] = (__bf16)W[idx];
}

// ---------------- layernorm: f32 row -> bf16 row ----------------
__global__ __launch_bounds__(256) void k_ln(const float* __restrict__ x,
                                            const float* __restrict__ g,
                                            const float* __restrict__ b,
                                            __bf16* __restrict__ out, int D) {
    const int row = blockIdx.x;
    const float* xr = x + (size_t)row * D;
    float s = 0.f, s2 = 0.f;
    for (int i = threadIdx.x; i < D; i += 256) { float v = xr[i]; s += v; s2 += v * v; }
    __shared__ float rs[256], rs2[256];
    rs[threadIdx.x] = s; rs2[threadIdx.x] = s2;
    __syncthreads();
    for (int off = 128; off > 0; off >>= 1) {
        if (threadIdx.x < off) { rs[threadIdx.x] += rs[threadIdx.x + off];
                                 rs2[threadIdx.x] += rs2[threadIdx.x + off]; }
        __syncthreads();
    }
    const float mean = rs[0] / (float)D;
    const float var  = rs2[0] / (float)D - mean * mean;
    const float inv  = rsqrtf(var + 1e-5f);
    for (int i = threadIdx.x; i < D; i += 256)
        out[(size_t)row * D + i] = (__bf16)((xr[i] - mean) * inv * g[i] + b[i]);
}

// ---------------- WMMA GEMM with TDM-staged, double-buffered LDS tiles -----------------
// C[M,N] = A[M,K](bf16) @ Wt[N,K]^T + bias ; block tile 128x128, 8 waves (4M x 2N),
// wave tile 32x64. Per K-slab (32), each wave DMAs one 32x32-element slice:
//   waves 0..3 -> A rows, waves 4..7 -> B rows.  Ping-pong LDS, s_wait_tensorcnt+barrier.
// MODE 0: outF = acc + bias
// MODE 1: outF = acc + bias + resid
// MODE 2: outB = gelu(acc+bias) + 0.1*sin(acc+bias)   (bf16)
template<int MODE>
__global__ __launch_bounds__(256) void k_gemm(const __bf16* __restrict__ A,
                                              const __bf16* __restrict__ Wt,
                                              const float* __restrict__ bias,
                                              const float* __restrict__ resid,
                                              float* __restrict__ outF,
                                              __bf16* __restrict__ outB,
                                              int M, int N, int K) {
    __shared__ __align__(128) __bf16 sA[2][128 * 32];
    __shared__ __align__(128) __bf16 sB[2][128 * 32];

    const int lane  = threadIdx.x & 31;
    const int wave  = threadIdx.x >> 5;
    const int blk_m = blockIdx.x * 128;
    const int blk_n = blockIdx.y * 128;
    const int m_loc = (wave >> 1) * 32;
    const int n_loc = (wave & 1) * 64;

    const bool isA = (wave < 4);
    const int  slc = isA ? wave : wave - 4;
    const __bf16* gbase = isA ? (A  + (size_t)(blk_m + slc * 32) * K)
                              : (Wt + (size_t)(blk_n + slc * 32) * K);

    f32x8 acc[2][4];
#pragma unroll
    for (int i = 0; i < 2; i++)
#pragma unroll
        for (int j = 0; j < 4; j++)
#pragma unroll
            for (int r = 0; r < 8; r++) acc[i][j][r] = 0.f;

    const int nslab = K >> 5;
    // prime slab 0
    tdm_tile_32x32(gbase, (isA ? &sA[0][slc * 32 * 32] : &sB[0][slc * 32 * 32]), K);

    for (int i = 0; i < nslab; i++) {
        __builtin_amdgcn_s_wait_tensorcnt(0);   // my slice of slab i landed in LDS
        __syncthreads();                        // everyone's slices visible
        const int cur = i & 1;
        if (i + 1 < nslab)                      // stream slab i+1 into the other buffer
            tdm_tile_32x32(gbase + (size_t)(i + 1) * 32,
                           (isA ? &sA[cur ^ 1][slc * 32 * 32]
                                : &sB[cur ^ 1][slc * 32 * 32]), K);

        const __bf16* a_s = &sA[cur][0];
        const __bf16* b_s = &sB[cur][0];
        bf16x16 a0 = load_a_frag(a_s, 32, m_loc,      0, lane);
        bf16x16 a1 = load_a_frag(a_s, 32, m_loc + 16, 0, lane);
        bf16x16 bfr[4];
#pragma unroll
        for (int j = 0; j < 4; j++)
            bfr[j] = load_b_frag(b_s, 32, n_loc + 16 * j, 0, lane);
#pragma unroll
        for (int j = 0; j < 4; j++) acc[0][j] = wmma_bf16(a0, bfr[j], acc[0][j]);
#pragma unroll
        for (int j = 0; j < 4; j++) acc[1][j] = wmma_bf16(a1, bfr[j], acc[1][j]);
    }

    // C layout: VGPR r, lanes 0-15 -> M=r ; lanes 16-31 -> M=r+8 ; N = lane&15
    const int cl = lane & 15;
    const int rg = (lane >> 4) << 3;
#pragma unroll
    for (int i = 0; i < 2; i++)
#pragma unroll
        for (int j = 0; j < 4; j++)
#pragma unroll
            for (int r = 0; r < 8; r++) {
                const int row = blk_m + m_loc + 16 * i + rg + r;
                const int col = blk_n + n_loc + 16 * j + cl;
                float v = acc[i][j][r] + bias[col];
                if constexpr (MODE == 1) v += resid[(size_t)row * N + col];
                if constexpr (MODE == 2) {
                    const float gel = 0.5f * v * (1.0f + erff(v * 0.70710678118654752f));
                    outB[(size_t)row * N + col] = (__bf16)(gel + 0.1f * __sinf(v));
                } else {
                    outF[(size_t)row * N + col] = v;
                }
            }
}

// ---------------- wave features: qn = l2norm(sin(f*t + p)) -> [B,H,T,32] bf16 ------------
__global__ __launch_bounds__(256) void k_wavefeat(const float* __restrict__ f,
                                                  const float* __restrict__ p,
                                                  __bf16* __restrict__ outn) {
    const int idx = blockIdx.x * 256 + threadIdx.x;   // tok * H + h
    const int h   = idx & 15;
    const int tok = idx >> 4;
    const int t   = tok & (SEQ - 1);
    const int b   = tok >> 11;
    const float* fp = f + (size_t)tok * (N_HEADS * N_WAVES) + h * N_WAVES;
    const float* pp = p + (size_t)tok * (N_HEADS * N_WAVES) + h * N_WAVES;
    float vals[N_WAVES]; float ss = 0.f;
#pragma unroll
    for (int w = 0; w < N_WAVES; w++) {
        const float s = __sinf(fp[w] * (float)t + pp[w]);
        vals[w] = s; ss += s * s;
    }
    const float inv = 1.f / fmaxf(sqrtf(ss), 1e-12f);
    __bf16* o = outn + ((size_t)(b * N_HEADS + h) * SEQ + t) * 32;
#pragma unroll
    for (int w = 0; w < N_WAVES; w++) o[w] = (__bf16)(vals[w] * inv);
#pragma unroll
    for (int w = N_WAVES; w < 32; w++) o[w] = (__bf16)0.f;   // zero-pad K 16->32
}

// ---------------- v transpose: [B,T,H*Dh] f32 -> [B,H,Dh,T] bf16 -----------------------
__global__ __launch_bounds__(256) void k_vtrans(const float* __restrict__ v,
                                                __bf16* __restrict__ vt) {
    const int idx = blockIdx.x * 256 + threadIdx.x;   // ((bh*64 + d)*SEQ + t)
    const int t  = idx & (SEQ - 1);
    const int bd = idx >> 11;
    const int d  = bd & 63;
    const int bh = bd >> 6;
    const int b = bh >> 4, h = bh & 15;
    vt[idx] = (__bf16)(v[((size_t)(b * SEQ + t)) * D_MODEL + h * D_HEAD + d]);
}

// ---------------- attention: one wave per 16 query rows ------------------------------
__global__ __launch_bounds__(32) void k_attn(const __bf16* __restrict__ qn,
                                             const __bf16* __restrict__ kn,
                                             const __bf16* __restrict__ vt,
                                             const float* __restrict__ iscale,
                                             __bf16* __restrict__ attn_out) {
    const int nblk = SEQ >> 4;
    const int bh = blockIdx.x / nblk;
    const int m0 = (blockIdx.x % nblk) << 4;
    const int h = bh & 15, b = bh >> 4;
    const int lane = threadIdx.x;
    const __bf16* q = qn + (size_t)bh * SEQ * 32;
    const __bf16* k = kn + (size_t)bh * SEQ * 32;
    const __bf16* v = vt + (size_t)bh * D_HEAD * SEQ;
    const float scale = iscale[h];

    __shared__ __align__(32) __bf16 s_attn[16 * 32];

    const bf16x16 aq = load_a_frag(q, 32, m0, 0, lane);  // q tile, K padded to 32
    f32x8 acc[4];
#pragma unroll
    for (int j = 0; j < 4; j++)
#pragma unroll
        for (int r = 0; r < 8; r++) acc[j][r] = 0.f;

    const int cl = lane & 15;
    const int rg = (lane >> 4) << 3;

    for (int s0 = 0; s0 <= m0 + 15; s0 += 32) {
        // two 16-wide score tiles -> bf16 attn tile [16 x 32] in LDS
#pragma unroll
        for (int st = 0; st < 2; st++) {
            const int sb = s0 + 16 * st;
            f32x8 sc;
#pragma unroll
            for (int r = 0; r < 8; r++) sc[r] = 0.f;
            const bf16x16 bk = load_b_frag(k, 32, sb, 0, lane);
            sc = wmma_bf16(aq, bk, sc);
#pragma unroll
            for (int r = 0; r < 8; r++) {
                const int tr = rg + r;            // row inside tile
                const int t  = m0 + tr;
                const int s  = sb + cl;
                float val = sc[r] * scale;
                if (s > t) val = 0.f;             // causal mask
                val *= rsqrtf((float)(t + 1));    // / sqrt(num_attended)
                s_attn[tr * 32 + 16 * st + cl] = (__bf16)val;
            }
        }
        __syncthreads();
        const bf16x16 aa = load_a_frag(s_attn, 32, 0, 0, lane);
#pragma unroll
        for (int j = 0; j < 4; j++) {
            const bf16x16 bv = load_b_frag(v, SEQ, 16 * j, s0, lane);
            acc[j] = wmma_bf16(aa, bv, acc[j]);
        }
        __syncthreads();
    }

    // write [B,T,H*Dh] bf16
#pragma unroll
    for (int j = 0; j < 4; j++)
#pragma unroll
        for (int r = 0; r < 8; r++) {
            const int t = m0 + rg + r;
            const int d = h * D_HEAD + 16 * j + cl;
            attn_out[((size_t)(b * SEQ + t)) * D_MODEL + d] = (__bf16)acc[j][r];
        }
}

// ================================ host launcher ================================
extern "C" void kernel_launch(void* const* d_in, const int* in_sizes, int n_in,
                              void* d_out, int out_size, void* d_ws, size_t ws_size,
                              hipStream_t stream) {
    const float* x    = (const float*)d_in[0];
    const float* ln1g = (const float*)d_in[1];
    const float* ln1b = (const float*)d_in[2];
    const float* Wqf  = (const float*)d_in[3];  const float* bqf = (const float*)d_in[4];
    const float* Wkf  = (const float*)d_in[5];  const float* bkf = (const float*)d_in[6];
    const float* Wqp  = (const float*)d_in[7];  const float* bqp = (const float*)d_in[8];
    const float* Wkp  = (const float*)d_in[9];  const float* bkp = (const float*)d_in[10];
    const float* Wv   = (const float*)d_in[11]; const float* bv  = (const float*)d_in[12];
    const float* Wo   = (const float*)d_in[13]; const float* bo  = (const float*)d_in[14];
    const float* isc  = (const float*)d_in[15];
    const float* ln2g = (const float*)d_in[16];
    const float* ln2b = (const float*)d_in[17];
    const float* W1   = (const float*)d_in[18]; const float* b1  = (const float*)d_in[19];
    const float* W2   = (const float*)d_in[20]; const float* b2  = (const float*)d_in[21];

    char* ws = (char*)d_ws;
    const size_t MB = 1024ull * 1024ull;
    __bf16* wt_qf = (__bf16*)(ws + 0 * MB);          // 0.5 MB each
    __bf16* wt_kf = (__bf16*)(ws + 0 * MB + 524288);
    __bf16* wt_qp = (__bf16*)(ws + 1 * MB);
    __bf16* wt_kp = (__bf16*)(ws + 1 * MB + 524288);
    __bf16* wt_v  = (__bf16*)(ws + 2 * MB);          // 2 MB
    __bf16* wt_o  = (__bf16*)(ws + 4 * MB);          // 2 MB
    __bf16* wt_1  = (__bf16*)(ws + 6 * MB);          // 8 MB
    __bf16* wt_2  = (__bf16*)(ws + 14 * MB);         // 8 MB
    __bf16* xn    = (__bf16*)(ws + 22 * MB);         // 8 MB (reused as LN2 out)
    float*  qf    = (float*)(ws + 30 * MB);          // 4 MB each
    float*  kf    = (float*)(ws + 34 * MB);
    float*  qp    = (float*)(ws + 38 * MB);
    float*  kp    = (float*)(ws + 42 * MB);
    float*  vf    = (float*)(ws + 46 * MB);          // 16 MB
    __bf16* qnb   = (__bf16*)(ws + 62 * MB);         // 8 MB
    __bf16* knb   = (__bf16*)(ws + 70 * MB);         // 8 MB
    __bf16* vtb   = (__bf16*)(ws + 78 * MB);         // 8 MB
    __bf16* aob   = (__bf16*)(ws + 86 * MB);         // 8 MB
    float*  x1    = (float*)(ws + 94 * MB);          // 16 MB
    __bf16* hact  = (__bf16*)(ws + 30 * MB);         // 32 MB, reuses dead proj bufs
    __bf16* hin   = xn;

    const dim3 B256(256);

    // weights -> bf16 [N,K]
    k_cvtw<<<(1024 * 256) / 256, B256, 0, stream>>>(Wqf, wt_qf, 1024, 256);
    k_cvtw<<<(1024 * 256) / 256, B256, 0, stream>>>(Wkf, wt_kf, 1024, 256);
    k_cvtw<<<(1024 * 256) / 256, B256, 0, stream>>>(Wqp, wt_qp, 1024, 256);
    k_cvtw<<<(1024 * 256) / 256, B256, 0, stream>>>(Wkp, wt_kp, 1024, 256);
    k_cvtw<<<(1024 * 1024) / 256, B256, 0, stream>>>(Wv, wt_v, 1024, 1024);
    k_cvtw<<<(1024 * 1024) / 256, B256, 0, stream>>>(Wo, wt_o, 1024, 1024);
    k_cvtw<<<(1024 * 4096) / 256, B256, 0, stream>>>(W1, wt_1, 1024, 4096);
    k_cvtw<<<(4096 * 1024) / 256, B256, 0, stream>>>(W2, wt_2, 4096, 1024);

    // LN1 -> bf16 activations
    k_ln<<<NTOK, B256, 0, stream>>>(x, ln1g, ln1b, xn, D_MODEL);

    // projections
    k_gemm<0><<<dim3(32, 2), B256, 0, stream>>>(xn, wt_qf, bqf, nullptr, qf, nullptr, NTOK, 256, 1024);
    k_gemm<0><<<dim3(32, 2), B256, 0, stream>>>(xn, wt_kf, bkf, nullptr, kf, nullptr, NTOK, 256, 1024);
    k_gemm<0><<<dim3(32, 2), B256, 0, stream>>>(xn, wt_qp, bqp, nullptr, qp, nullptr, NTOK, 256, 1024);
    k_gemm<0><<<dim3(32, 2), B256, 0, stream>>>(xn, wt_kp, bkp, nullptr, kp, nullptr, NTOK, 256, 1024);
    k_gemm<0><<<dim3(32, 8), B256, 0, stream>>>(xn, wt_v, bv, nullptr, vf, nullptr, NTOK, 1024, 1024);

    // wave features + v transpose
    k_wavefeat<<<(NTOK * N_HEADS) / 256, B256, 0, stream>>>(qf, qp, qnb);
    k_wavefeat<<<(NTOK * N_HEADS) / 256, B256, 0, stream>>>(kf, kp, knb);
    k_vtrans<<<(BATCH * N_HEADS * D_HEAD * SEQ) / 256, B256, 0, stream>>>(vf, vtb);

    // attention
    k_attn<<<BATCH * N_HEADS * (SEQ / 16), dim3(32), 0, stream>>>(qnb, knb, vtb, isc, aob);

    // out proj + residual
    k_gemm<1><<<dim3(32, 8), B256, 0, stream>>>(aob, wt_o, bo, x, x1, nullptr, NTOK, 1024, 1024);

    // LN2 + FFN
    k_ln<<<NTOK, B256, 0, stream>>>(x1, ln2g, ln2b, hin, D_MODEL);
    k_gemm<2><<<dim3(32, 32), B256, 0, stream>>>(hin, wt_1, b1, nullptr, nullptr, hact, NTOK, D_FF, 1024);
    k_gemm<1><<<dim3(32, 8), B256, 0, stream>>>(hact, wt_2, b2, x1, (float*)d_out, nullptr, NTOK, 1024, D_FF);
}
